// SelfAttention_53068615910345
// MI455X (gfx1250) — matrix operand order repaired and checked by reading, not compiled, not run on hardware
//
#include <hip/hip_runtime.h>

#define EMBED   1024
#define SEQ     2048
#define BATCH   2
#define NHEADS  16
#define HD      64
#define MROWS   (BATCH * SEQ)     // 4096

typedef __attribute__((ext_vector_type(16))) __bf16 v16bf;
typedef __attribute__((ext_vector_type(8)))  float  v8f;

static __device__ __forceinline__ v8f wmma_bf16(v16bf a, v16bf b, v8f c) {
    // D = A(16x32 bf16) x B(32x16 bf16) + C(16x16 f32)
    return __builtin_amdgcn_wmma_f32_16x16x32_bf16(
        false, a, false, b, (short)0, c, false, false);
}

// A-fragment K index for element e of v16bf (16-bit A 16x32 layout, ISA 7.12.2)
static __device__ __forceinline__ int a_kidx(int e, int kb) {
    return (e < 8) ? (kb + e) : (16 + kb + (e - 8));
}

// 16-byte async global -> LDS copy (ASYNCcnt-tracked, ISA 10.7 / 15.18.3)
static __device__ __forceinline__ void async_copy16(const __bf16* g, const __bf16* l) {
    unsigned lds = (unsigned)(unsigned long long)l;   // flat low 32 bits = LDS addr
    asm volatile("global_load_async_to_lds_b128 %0, %1, off"
                 :: "v"(lds), "v"(g) : "memory");
}
static __device__ __forceinline__ void wait_async0() {
    asm volatile("s_wait_asynccnt 0" ::: "memory");
}

// ---------------------------------------------------------------------------
// fp32 -> bf16 conversion
// ---------------------------------------------------------------------------
__global__ __launch_bounds__(256)
void convert_bf16(const float* __restrict__ in, __bf16* __restrict__ out, int n) {
    int i = blockIdx.x * blockDim.x + threadIdx.x;
    if (i < n) out[i] = (__bf16)in[i];
}

// ---------------------------------------------------------------------------
// Tiled GEMM: C[64x64 tile] = A[M,K] @ B[K,N] + bias
// Double-buffered async-LDS staging; K-step 32; 4 waves x (16 rows x 64 cols).
// MODE 0: QKV epilogue -> scatter bf16 into Q(,scaled)/K/V [B,H,S,D]
// MODE 1: plain epilogue -> fp32 C
// ---------------------------------------------------------------------------
template <int MODE>
__global__ __launch_bounds__(128)
void gemm_bf16(const __bf16* __restrict__ A, const __bf16* __restrict__ Bm,
               const float* __restrict__ bias, int Kdim, int Ncols,
               __bf16* __restrict__ Qb, __bf16* __restrict__ Kb,
               __bf16* __restrict__ Vb, float* __restrict__ Cout)
{
    __shared__ __align__(16) __bf16 As[2][64][40];   // 64 rows x 32 K (+pad, 80B stride)
    __shared__ __align__(16) __bf16 Bs[2][32][72];   // 32 K x 64 cols (+pad, 144B stride)

    const int tid  = threadIdx.x;
    const int lane = tid & 31;
    const int w    = tid >> 5;
    const int hi   = lane >> 4;      // lane half
    const int nn   = lane & 15;
    const int kb   = hi * 8;         // A-frag K base
    const int kb2  = hi * 16;        // unused; B frag also uses kb
    const int m0   = blockIdx.y * 64;
    const int n0   = blockIdx.x * 64;

    // issue async loads of one K-step tile pair into buffer bb
    auto issue_k = [&](int k0, int bb) {
        #pragma unroll
        for (int j = 0; j < 2; ++j) {
            const int l = tid + j * 128;               // 0..255
            const int ia = l >> 2, ca = l & 3;         // A: 64 rows x 4 chunks
            async_copy16(A + (size_t)(m0 + ia) * Kdim + k0 + ca * 8,
                         &As[bb][ia][ca * 8]);
            const int ib = l >> 3, cb = l & 7;         // B: 32 rows x 8 chunks
            async_copy16(Bm + (size_t)(k0 + ib) * Ncols + n0 + cb * 8,
                         &Bs[bb][ib][cb * 8]);
        }
    };

    v8f c[4] = {};
    issue_k(0, 0);

    for (int k0 = 0; k0 < Kdim; k0 += 32) {
        const int bb = (k0 >> 5) & 1;
        wait_async0();
        __syncthreads();
        if (k0 + 32 < Kdim) issue_k(k0 + 32, bb ^ 1);  // overlap next fetch

        v16bf a;
        #pragma unroll
        for (int e = 0; e < 16; ++e)
            a[e] = As[bb][w * 16 + nn][a_kidx(e, kb)];

        #pragma unroll
        for (int t = 0; t < 4; ++t) {
            v16bf bf;
            #pragma unroll
            for (int e = 0; e < 16; ++e)
                bf[e] = Bs[bb][a_kidx(e, kb)][t * 16 + nn];
            c[t] = wmma_bf16(a, bf, c[t]);
        }
        // no trailing barrier: buffer bb is only overwritten after the next
        // top-of-loop barrier, which every wave reaches after these reads.
    }

    #pragma unroll
    for (int t = 0; t < 4; ++t) {
        const int ngl = n0 + t * 16 + nn;
        const float bv = bias[ngl];
        #pragma unroll
        for (int r = 0; r < 8; ++r) {
            const int mgl = m0 + w * 16 + hi * 8 + r;
            const float val = c[t][r] + bv;
            if (MODE == 0) {
                const int which = ngl >> 10;        // 0=q 1=k 2=v
                const int e = ngl & 1023;
                const int h = e >> 6, d = e & 63;
                const int bi = mgl >> 11, s = mgl & 2047;
                const size_t idx = ((size_t)(bi * NHEADS + h) * SEQ + s) * HD + d;
                if (which == 0)      Qb[idx] = (__bf16)(val * 0.125f); // fold 1/sqrt(64)
                else if (which == 1) Kb[idx] = (__bf16)val;
                else                 Vb[idx] = (__bf16)val;
            } else {
                Cout[(size_t)mgl * Ncols + ngl] = val;
            }
        }
    }
}

// ---------------------------------------------------------------------------
// Causal flash attention. Grid: (S/64, B*H). Block: 128 (4 waves).
// Wave owns 16 q rows; streams 64-key tiles (double-buffered async-LDS);
// online softmax in fp32.
// ---------------------------------------------------------------------------
__global__ __launch_bounds__(128)
void attn_kernel(const __bf16* __restrict__ Qg, const __bf16* __restrict__ Kg,
                 const __bf16* __restrict__ Vg, __bf16* __restrict__ Og)
{
    __shared__ __align__(16) __bf16 Kt[2][64][72];
    __shared__ __align__(16) __bf16 Vt[2][64][72];
    __shared__ __align__(16) __bf16 Pt[4][16][72];

    const int tid  = threadIdx.x;
    const int lane = tid & 31;
    const int w    = tid >> 5;
    const int hi   = lane >> 4;
    const int nn   = lane & 15;
    const int kb   = hi * 8;
    const int kb2  = hi * 16;
    const int q0   = blockIdx.x * 64;
    const int bh   = blockIdx.y;              // b*16 + h
    const int b    = bh >> 4, h = bh & 15;

    const __bf16* Qp = Qg + (size_t)bh * SEQ * HD;
    const __bf16* Kp = Kg + (size_t)bh * SEQ * HD;
    const __bf16* Vp = Vg + (size_t)bh * SEQ * HD;

    // issue async loads of K/V tile kt into buffer bb (64x64 bf16 each)
    auto issue_tile = [&](int kt, int bb) {
        const int kbase = kt * 64;
        #pragma unroll
        for (int j = 0; j < 4; ++j) {
            const int l = tid + j * 128;      // 0..511
            const int i = l >> 3, cc = l & 7; // row, 16B chunk
            async_copy16(Kp + (size_t)(kbase + i) * HD + cc * 8, &Kt[bb][i][cc * 8]);
            async_copy16(Vp + (size_t)(kbase + i) * HD + cc * 8, &Vt[bb][i][cc * 8]);
        }
    };

    // Q A-fragments for this wave's 16 rows (d split into two 32-chunks)
    v16bf aq[2];
    #pragma unroll
    for (int cc = 0; cc < 2; ++cc)
        #pragma unroll
        for (int e = 0; e < 16; ++e)
            aq[cc][e] = Qp[(size_t)(q0 + w * 16 + nn) * HD + cc * 32 + a_kidx(e, kb)];

    float mrun[8], lrun[8];
    v8f o[4] = {};
    #pragma unroll
    for (int r = 0; r < 8; ++r) { mrun[r] = -1e30f; lrun[r] = 0.f; }

    const int nkt = blockIdx.x + 1;           // causal: ktiles 0..qtile
    issue_tile(0, 0);

    for (int kt = 0; kt < nkt; ++kt) {
        const int kbase = kt * 64;
        const int bb = kt & 1;
        wait_async0();
        __syncthreads();
        if (kt + 1 < nkt) issue_tile(kt + 1, bb ^ 1);   // overlap next fetch

        // scores: S = Q @ K^T   (B element (kk,n) = K[key=n][d=kk])
        v8f sc[4] = {};
        #pragma unroll
        for (int cc = 0; cc < 2; ++cc) {
            #pragma unroll
            for (int t = 0; t < 4; ++t) {
                v16bf bk;
                #pragma unroll
                for (int e = 0; e < 16; ++e)
                    bk[e] = Kt[bb][t * 16 + nn][cc * 32 + a_kidx(e, kb)];
                sc[t] = wmma_bf16(aq[cc], bk, sc[t]);
            }
        }

        // causal mask + online softmax (rows live across 16-lane halves)
        #pragma unroll
        for (int r = 0; r < 8; ++r) {
            const int qi = q0 + w * 16 + hi * 8 + r;
            float rowmax = -1e30f;
            #pragma unroll
            for (int t = 0; t < 4; ++t) {
                const int kj = kbase + t * 16 + nn;
                float v = sc[t][r];
                if (kj > qi) v = -1e30f;
                sc[t][r] = v;
                rowmax = fmaxf(rowmax, v);
            }
            for (int off = 1; off < 16; off <<= 1)
                rowmax = fmaxf(rowmax, __shfl_xor(rowmax, off, 32));
            const float mnew  = fmaxf(mrun[r], rowmax);
            const float scale = __expf(mrun[r] - mnew);
            float rsum = 0.f;
            #pragma unroll
            for (int t = 0; t < 4; ++t) {
                const float p = __expf(sc[t][r] - mnew);
                sc[t][r] = p;
                rsum += p;
            }
            for (int off = 1; off < 16; off <<= 1)
                rsum += __shfl_xor(rsum, off, 32);
            lrun[r] = lrun[r] * scale + rsum;
            mrun[r] = mnew;
            #pragma unroll
            for (int t = 0; t < 4; ++t)
                o[t][r] *= scale;
        }

        // P (D-layout) -> LDS -> A-fragment layout; per-wave region, in-order DS
        #pragma unroll
        for (int t = 0; t < 4; ++t)
            #pragma unroll
            for (int r = 0; r < 8; ++r)
                Pt[w][hi * 8 + r][t * 16 + nn] = (__bf16)sc[t][r];

        // O += P @ V
        #pragma unroll
        for (int cc = 0; cc < 2; ++cc) {
            v16bf ap;
            #pragma unroll
            for (int e = 0; e < 16; ++e)
                ap[e] = Pt[w][nn][cc * 32 + a_kidx(e, kb)];
            #pragma unroll
            for (int t = 0; t < 4; ++t) {
                v16bf bv;
                #pragma unroll
                for (int e = 0; e < 16; ++e)
                    bv[e] = Vt[bb][cc * 32 + a_kidx(e, kb)][t * 16 + nn];
                o[t] = wmma_bf16(ap, bv, o[t]);
            }
        }
        // no trailing barrier: buffer bb is only overwritten after the next
        // top-of-loop barrier, which every wave reaches after these reads.
    }

    // finalize: O /= l ; write [B, S, h*64+d] bf16
    #pragma unroll
    for (int r = 0; r < 8; ++r) {
        const float inv = 1.0f / lrun[r];
        const int srow  = q0 + w * 16 + hi * 8 + r;
        #pragma unroll
        for (int t = 0; t < 4; ++t) {
            const int d = t * 16 + nn;
            Og[((size_t)(b * SEQ + srow)) * EMBED + h * HD + d] = (__bf16)(o[t][r] * inv);
        }
    }
}

// ---------------------------------------------------------------------------
extern "C" void kernel_launch(void* const* d_in, const int* in_sizes, int n_in,
                              void* d_out, int out_size, void* d_ws, size_t ws_size,
                              hipStream_t stream) {
    const float* x     = (const float*)d_in[0];
    const float* W_qkv = (const float*)d_in[1];
    const float* b_qkv = (const float*)d_in[2];
    const float* W_out = (const float*)d_in[3];
    const float* b_out = (const float*)d_in[4];
    float* out = (float*)d_out;

    __bf16* wsb    = (__bf16*)d_ws;
    __bf16* xbf    = wsb;                                   // 4096*1024
    __bf16* wqkvbf = xbf    + (size_t)MROWS * EMBED;        // 1024*3072
    __bf16* woutbf = wqkvbf + (size_t)EMBED * 3 * EMBED;    // 1024*1024
    __bf16* Qb     = woutbf + (size_t)EMBED * EMBED;        // [B,H,S,D]
    __bf16* Kb     = Qb + (size_t)BATCH * NHEADS * SEQ * HD;
    __bf16* Vb     = Kb + (size_t)BATCH * NHEADS * SEQ * HD;
    __bf16* Ob     = Vb + (size_t)BATCH * NHEADS * SEQ * HD; // [B,S,E]

    const int n1 = MROWS * EMBED;
    const int n2 = EMBED * 3 * EMBED;
    const int n3 = EMBED * EMBED;
    convert_bf16<<<(n1 + 255) / 256, 256, 0, stream>>>(x, xbf, n1);
    convert_bf16<<<(n2 + 255) / 256, 256, 0, stream>>>(W_qkv, wqkvbf, n2);
    convert_bf16<<<(n3 + 255) / 256, 256, 0, stream>>>(W_out, woutbf, n3);

    dim3 g1(3 * EMBED / 64, MROWS / 64);   // (48, 64)
    gemm_bf16<0><<<g1, 128, 0, stream>>>(xbf, wqkvbf, b_qkv, EMBED, 3 * EMBED,
                                         Qb, Kb, Vb, nullptr);

    dim3 g2(SEQ / 64, BATCH * NHEADS);     // (32, 32)
    attn_kernel<<<g2, 128, 0, stream>>>(Qb, Kb, Vb, Ob);

    dim3 g3(EMBED / 64, MROWS / 64);       // (16, 64)
    gemm_bf16<1><<<g3, 128, 0, stream>>>(Ob, woutbf, b_out, EMBED, EMBED,
                                         nullptr, nullptr, nullptr, out);
}
